// LSHSelfCrossAttention_1494648619505
// MI455X (gfx1250) — matrix operand order repaired and checked by reading, not compiled, not run on hardware
//
#include <hip/hip_runtime.h>

// ---------------- problem constants ----------------
constexpr int Bc   = 2;
constexpr int Sc   = 4096;
constexpr int Hc   = 16;
constexpr int Dc   = 64;
constexpr int HIDc = 1024;
constexpr int NHASH = 2;
constexpr int NBUCK = 128;
constexpr int NSc  = NHASH * Sc;          // 8192
constexpr int NCc  = NSc / 64;            // 128 chunks
constexpr int Nc   = Hc * Dc;             // 1024 (proj N)
constexpr float SELF_MASK = -1e5f;

// ---------------- WMMA types ----------------
typedef __attribute__((ext_vector_type(16))) __bf16 v16bf;
typedef __attribute__((ext_vector_type(8)))  float  v8f;

union BF16x16 {
    v16bf v;
    unsigned short u[16];
    unsigned int   w[8];
};

// native bf16 converts -> hardware v_cvt ops instead of manual rounding
__device__ __forceinline__ unsigned short f2bf(float f) {
    __bf16 h = (__bf16)f;
    return __builtin_bit_cast(unsigned short, h);
}
__device__ __forceinline__ float bf2f(unsigned short b) {
    return (float)__builtin_bit_cast(__bf16, b);
}

// ---------------- device RNG (stand-in for jax threefry) ----------------
__device__ __forceinline__ unsigned int pcg_hash(unsigned int v) {
    v = v * 747796405u + 2891336453u;
    unsigned int w = ((v >> ((v >> 28u) + 4u)) ^ v) * 277803737u;
    return (w >> 22u) ^ w;
}
__device__ __forceinline__ float hash_normal(unsigned int idx) {
    unsigned int a = pcg_hash(idx * 2u + 1u);
    unsigned int b = pcg_hash(idx * 2u + 2u);
    float u1 = ((float)a + 1.0f) * 2.3283064e-10f;   // (0,1]
    float u2 = (float)b * 2.3283064e-10f;
    return sqrtf(-2.0f * __logf(u1)) * __cosf(6.2831853f * u2);
}

// ---------------- 1. rotations, bf16, transposed layout [H][nr=128][d=64] ----------------
__global__ void rot_kernel(unsigned short* __restrict__ rotbf) {
    int i = blockIdx.x * 256 + threadIdx.x;           // < H*128*64 = 131072
    rotbf[i] = f2bf(hash_normal((unsigned int)i));
}

// ---------------- 2. projection GEMM (fp32 in, bf16 WMMA, bf16 out [B,H,S,D]) ----------------
__global__ __launch_bounds__(256) void proj_kernel(
    const float* __restrict__ dec, const float* __restrict__ hid,
    const float* __restrict__ wqk, const float* __restrict__ wv,
    unsigned short* __restrict__ qb, unsigned short* __restrict__ kb,
    unsigned short* __restrict__ vb)
{
    __shared__ unsigned short Asub[128 * 32];   // 8 KB
    __shared__ unsigned short Wt[16 * 32];      // 1 KB (transposed: [n][k])

    const int z = blockIdx.z;
    const float* A = (z == 0) ? dec : hid;
    const float* W = (z == 2) ? wv : wqk;
    unsigned short* out = (z == 0) ? qb : (z == 1 ? kb : vb);

    const int m0 = blockIdx.x * 128;
    const int n0 = blockIdx.y * 16;
    const int tid = threadIdx.x, lane = tid & 31, wv32 = tid >> 5;
    const int half = lane >> 4;

    v8f acc = {};
    const int arow = tid >> 1;
    const int acol0 = (tid & 1) * 16;

    for (int kk = 0; kk < HIDc; kk += 32) {
        // stage A tile (128x32) as bf16; prefetch next k-tile of this row
        {
            const float* ag = A + (size_t)(m0 + arow) * HIDc + kk + acol0;
            if (kk + 32 < HIDc) __builtin_prefetch(ag + 32, 0, 1);
            unsigned short* ad = Asub + arow * 32 + acol0;
#pragma unroll
            for (int i = 0; i < 16; i++) ad[i] = f2bf(ag[i]);
        }
        // stage W tile transposed: Wt[n][k] = W[kk+k][n0+n]
        {
            int kI = tid & 31, nb = tid >> 5;
            Wt[nb * 32 + kI]       = f2bf(W[(size_t)(kk + kI) * Nc + n0 + nb]);
            Wt[(nb + 8) * 32 + kI] = f2bf(W[(size_t)(kk + kI) * Nc + n0 + nb + 8]);
        }
        __syncthreads();

        BF16x16 af, bf;
        const unsigned int* ar = (const unsigned int*)(Asub + (wv32 * 16 + (lane & 15)) * 32);
#pragma unroll
        for (int j = 0; j < 4; j++) {
            af.w[j]     = ar[4 * half + j];
            af.w[4 + j] = ar[8 + 4 * half + j];
        }
        const unsigned int* br = (const unsigned int*)(Wt + (lane & 15) * 32);
#pragma unroll
        for (int j = 0; j < 8; j++) bf.w[j] = br[4 * half + (j & 3) + 8 * (j >> 2)];

        acc = __builtin_amdgcn_wmma_f32_16x16x32_bf16(false, af.v, false, bf.v,
                                                      (short)0, acc, false, false);
        __syncthreads();
    }

    // C layout: lane n = lane&15; VGPR r -> m = r + 8*(lane>=16)
    const int n = lane & 15;
    const int col = n0 + n;
    const int h = col >> 6, d = col & 63;
    const int rowbase = m0 + wv32 * 16 + 8 * half;
#pragma unroll
    for (int r = 0; r < 8; r++) {
        int rr = rowbase + r;
        int b = rr >> 12, s = rr & 4095;
        out[(((size_t)b * Hc + h) * Sc + s) * Dc + d] = f2bf(acc[r]);
    }
}

// ---------------- 3. LSH hashing via WMMA (raw q/k, before K-norm) ----------------
// rotated[s, nr] = sum_d x[s,d] * rot[nr,d]  -> argmax over [x, -x]
__global__ __launch_bounds__(256) void hash_kernel(
    const unsigned short* __restrict__ qb, const unsigned short* __restrict__ kb,
    const unsigned short* __restrict__ rotbf, int* __restrict__ buckets)
{
    __shared__ unsigned short qsb[64 * 64];    //  8 KB: 64 rows of x
    __shared__ unsigned short rotT[128 * 64];  // 16 KB: [nr][d]
    __shared__ float rotd[64 * 128];           // 32 KB: scores

    const int bh = blockIdx.x, h = bh & (Hc - 1);
    const int s0 = blockIdx.y * 64;
    const int z = blockIdx.z;
    const int tid = threadIdx.x, lane = tid & 31, w = tid >> 5;
    const int half = lane >> 4;
    const unsigned short* src = (z ? kb : qb) + ((size_t)bh * Sc + s0) * Dc;

    for (int e = tid; e < 64 * 32; e += 256)
        ((unsigned int*)qsb)[e] = ((const unsigned int*)src)[e];
    for (int e = tid; e < 128 * 32; e += 256)
        ((unsigned int*)rotT)[e] = ((const unsigned int*)(rotbf + (size_t)h * 128 * 64))[e];
    __syncthreads();

    {
        const int qm  = w >> 1;
        const int cn0 = (w & 1) * 4;
        const unsigned int* arow = (const unsigned int*)(qsb + (qm * 16 + (lane & 15)) * 64);
        for (int cn = cn0; cn < cn0 + 4; cn++) {
            v8f acc = {};
            const unsigned int* brow = (const unsigned int*)(rotT + (cn * 16 + (lane & 15)) * 64);
#pragma unroll
            for (int kk = 0; kk < 64; kk += 32) {
                BF16x16 af, bf;
                int kb32 = kk >> 1;
#pragma unroll
                for (int j = 0; j < 4; j++) {
                    af.w[j]     = arow[kb32 + 4 * half + j];
                    af.w[4 + j] = arow[kb32 + 8 + 4 * half + j];
                }
#pragma unroll
                for (int j = 0; j < 8; j++) bf.w[j] = brow[kb32 + 4 * half + (j & 3) + 8 * (j >> 2)];
                acc = __builtin_amdgcn_wmma_f32_16x16x32_bf16(false, af.v, false, bf.v,
                                                              (short)0, acc, false, false);
            }
            int n = lane & 15, mb = 8 * half;
#pragma unroll
            for (int r = 0; r < 8; r++)
                rotd[(qm * 16 + mb + r) * 128 + cn * 16 + n] = acc[r];
        }
    }
    __syncthreads();

    if (tid < 64) {
        int* dst = buckets + ((size_t)z * (Bc * Hc) + bh) * NSc;
        const float* rr = rotd + tid * 128;
        for (int n = 0; n < NHASH; n++) {
            float bp = -3.4e38f, bn = -3.4e38f;
            int ap = 0, an = 0;
            for (int r = 0; r < 64; r++) {
                float v = rr[n * 64 + r];
                if (v > bp)  { bp = v;  ap = r; }
                if (-v > bn) { bn = -v; an = r; }
            }
            int arg = (bp >= bn) ? ap : (64 + an);   // argmax over [x, -x], first-wins
            dst[n * Sc + s0 + tid] = arg + n * NBUCK;
        }
    }
}

// ---------------- 4. in-LDS bitonic sort per (b,h) ----------------
__global__ __launch_bounds__(1024) void sort_kernel(
    const int* __restrict__ buckets, int* __restrict__ sq,
    int* __restrict__ sk, int* __restrict__ undo)
{
    __shared__ unsigned long long keys[NSc];        // 64 KB
    const int bh = blockIdx.x;
    const int which = blockIdx.y;                   // 0 = q, 1 = k
    const int* bsrc = buckets + ((size_t)which * (Bc * Hc) + bh) * NSc;

    for (int i = threadIdx.x; i < NSc; i += 1024) {
        unsigned long long scaled =
            (unsigned long long)((unsigned)bsrc[i] * Sc + (i & (Sc - 1)));
        keys[i] = (scaled << 13) | (unsigned)i;     // unique keys; stability moot
    }
    __syncthreads();

    for (unsigned k = 2; k <= (unsigned)NSc; k <<= 1) {
        for (unsigned j = k >> 1; j > 0; j >>= 1) {
            for (unsigned i = threadIdx.x; i < (unsigned)NSc; i += 1024) {
                unsigned ixj = i ^ j;
                if (ixj > i) {
                    unsigned long long a = keys[i], b = keys[ixj];
                    bool up = ((i & k) == 0);
                    if ((a > b) == up) { keys[i] = b; keys[ixj] = a; }
                }
            }
            __syncthreads();
        }
    }

    for (int p = threadIdx.x; p < NSc; p += 1024) {
        int idx = (int)(keys[p] & 0x1FFFULL);
        if (which == 0) {
            sq[(size_t)bh * NSc + p] = idx;
        } else {
            sk[(size_t)bh * NSc + p] = idx;
            undo[(size_t)bh * NSc + idx] = p;       // inverse permutation
        }
    }
}

// ---------------- 5. K row-norm: x * rsqrt(mean(x^2)+eps) / sqrt(D) ----------------
__global__ __launch_bounds__(256) void norm_k_kernel(unsigned short* __restrict__ kb) {
    const int tid = threadIdx.x, lane = tid & 31, w = tid >> 5;
    const size_t row = (size_t)blockIdx.x * 8 + w;  // < B*H*S
    unsigned int* p = (unsigned int*)(kb + row * Dc);
    unsigned int pk = p[lane];
    float f0 = bf2f((unsigned short)(pk & 0xFFFFu));
    float f1 = bf2f((unsigned short)(pk >> 16));
    float ss = f0 * f0 + f1 * f1;
#pragma unroll
    for (int off = 16; off > 0; off >>= 1) ss += __shfl_xor(ss, off, 32);
    float scale = rsqrtf(ss * (1.0f / 64.0f) + 1e-6f) * 0.125f;
    p[lane] = (unsigned int)f2bf(f0 * scale) | ((unsigned int)f2bf(f1 * scale) << 16);
}

// ---------------- 6. chunked LSH attention (WMMA) ----------------
__global__ __launch_bounds__(256) void attn_kernel(
    const unsigned short* __restrict__ qb, const unsigned short* __restrict__ kb,
    const unsigned short* __restrict__ vb, const int* __restrict__ sq,
    const int* __restrict__ sk, float* __restrict__ outws, float* __restrict__ logws)
{
    __shared__ unsigned short qs[64 * 64];     //  8 KB gathered queries
    __shared__ unsigned short ks[128 * 64];    // 16 KB gathered keys (prev+cur chunk)
    __shared__ unsigned short vsT[64 * 128];   // 16 KB gathered values, transposed [d][j]
    __shared__ float dots[64 * 128];           // 32 KB scores -> probs
    __shared__ int qpos[64];
    __shared__ int kvmask[128];                // self-mask positions (from sorted Q idx)
    __shared__ int kvsrc[128];                 // gather positions   (from sorted K idx)
    __shared__ float lse[64];

    const int c  = blockIdx.x;
    const int bh = blockIdx.y;
    const int tid = threadIdx.x, lane = tid & 31, w = tid >> 5;
    const int half = lane >> 4;
    const size_t bhNS = (size_t)bh * NSc;
    const size_t goff = (size_t)bh * Sc * Dc;

    if (tid < 64) qpos[tid] = sq[bhNS + c * 64 + tid] & (Sc - 1);
    if (tid < 128) {
        int jj = (tid < 64) ? (((c + NCc - 1) & (NCc - 1)) * 64 + tid)
                            : (c * 64 + tid - 64);
        kvmask[tid] = sq[bhNS + jj] & (Sc - 1);
        kvsrc[tid]  = sk[bhNS + jj] & (Sc - 1);
    }
    __syncthreads();

    // gather rows into LDS (u32 granular for q/k; elementwise transpose for v)
    for (int e = tid; e < 64 * 16; e += 256) {
        int r = e >> 4, cu = e & 15;
        const unsigned int* src = (const unsigned int*)(qb + goff + (size_t)qpos[r] * Dc);
        ((unsigned int*)qs)[r * 32 + cu]      = src[cu];
        ((unsigned int*)qs)[r * 32 + 16 + cu] = src[16 + cu];
    }
    for (int e = tid; e < 128 * 32; e += 256) {
        int r = e >> 5, cu = e & 31;
        const unsigned int* src = (const unsigned int*)(kb + goff + (size_t)kvsrc[r] * Dc);
        ((unsigned int*)ks)[e] = src[cu];
    }
    for (int e = tid; e < 128 * 64; e += 256) {
        int j = e >> 6, d = e & 63;
        vsT[d * 128 + j] = vb[goff + (size_t)kvsrc[j] * Dc + d];
    }
    __syncthreads();

    // ---- dots = qs(64x64) . ks^T(64x128) via bf16 WMMA; 4 tiles/wave ----
    {
        const int qm  = w >> 1;
        const int cn0 = (w & 1) * 4;
        const unsigned int* arow = (const unsigned int*)(qs + (qm * 16 + (lane & 15)) * 64);
        for (int cn = cn0; cn < cn0 + 4; cn++) {
            v8f acc = {};
            const unsigned int* brow = (const unsigned int*)(ks + (cn * 16 + (lane & 15)) * 64);
#pragma unroll
            for (int kk = 0; kk < 64; kk += 32) {
                BF16x16 af, bf;
                int kb32 = kk >> 1;
#pragma unroll
                for (int j = 0; j < 4; j++) {
                    af.w[j]     = arow[kb32 + 4 * half + j];
                    af.w[4 + j] = arow[kb32 + 8 + 4 * half + j];
                }
#pragma unroll
                for (int j = 0; j < 8; j++) bf.w[j] = brow[kb32 + 4 * half + (j & 3) + 8 * (j >> 2)];
                acc = __builtin_amdgcn_wmma_f32_16x16x32_bf16(false, af.v, false, bf.v,
                                                              (short)0, acc, false, false);
            }
            int n = lane & 15, mb = 8 * half;
#pragma unroll
            for (int r = 0; r < 8; r++)
                dots[(qm * 16 + mb + r) * 128 + cn * 16 + n] = acc[r];
        }
    }
    __syncthreads();

    // ---- self-mask + logsumexp softmax ----
    for (int e = tid; e < 64 * 128; e += 256) {
        int i = e >> 7, j = e & 127;
        if (qpos[i] == kvmask[j]) dots[e] = SELF_MASK;
    }
    __syncthreads();
    if (tid < 64) {
        const float* dr = dots + tid * 128;
        float mx = -3.4e38f;
        for (int j = 0; j < 128; j++) mx = fmaxf(mx, dr[j]);
        float sum = 0.f;
        for (int j = 0; j < 128; j++) sum += __expf(dr[j] - mx);
        float L = mx + __logf(sum);
        lse[tid] = L;
        logws[bhNS + c * 64 + tid] = L;
    }
    __syncthreads();
    for (int e = tid; e < 64 * 128; e += 256)
        dots[e] = __expf(dots[e] - lse[e >> 7]);
    __syncthreads();

    // ---- out = probs(64x128) . vs(128x64) via bf16 WMMA; 2 tiles/wave ----
    for (int t = 2 * w; t < 2 * w + 2; t++) {
        const int qm = t >> 2, dn = t & 3;
        const float* prow = dots + (qm * 16 + (lane & 15)) * 128;
        const unsigned int* vrow =
            (const unsigned int*)(vsT + (dn * 16 + (lane & 15)) * 128);
        v8f acc = {};
#pragma unroll
        for (int kk = 0; kk < 128; kk += 32) {
            BF16x16 af, bf;
#pragma unroll
            for (int i = 0; i < 8; i++) {
                af.u[i]     = f2bf(prow[kk + 8 * half + i]);
                af.u[8 + i] = f2bf(prow[kk + 16 + 8 * half + i]);
            }
            int b0 = (kk >> 1) + 4 * half;
#pragma unroll
            for (int j = 0; j < 8; j++) bf.w[j] = vrow[b0 + (j & 3) + 8 * (j >> 2)];
            acc = __builtin_amdgcn_wmma_f32_16x16x32_bf16(false, af.v, false, bf.v,
                                                          (short)0, acc, false, false);
        }
        int n = lane & 15, mb = 8 * half;
        size_t obase = (bhNS + c * 64) * Dc;
#pragma unroll
        for (int r = 0; r < 8; r++)
            outws[obase + (size_t)(qm * 16 + mb + r) * Dc + dn * 16 + n] = acc[r];
    }
}

// ---------------- 7. unsort (undo_k) + hash-round merge + transpose store ----------------
__global__ __launch_bounds__(64) void merge_kernel(
    const float* __restrict__ outws, const float* __restrict__ logws,
    const int* __restrict__ undo, float* __restrict__ out)
{
    const int s = blockIdx.x, bh = blockIdx.y, d = threadIdx.x;
    const int b = bh >> 4, h = bh & 15;
    const size_t base = (size_t)bh * NSc;
    int u0 = undo[base + s];
    int u1 = undo[base + Sc + s];
    float l0 = logws[base + u0], l1 = logws[base + u1];
    float m = fmaxf(l0, l1);
    float e0 = __expf(l0 - m), e1 = __expf(l1 - m);
    float inv = 1.0f / (e0 + e1);
    float o = (e0 * outws[(base + u0) * Dc + d] +
               e1 * outws[(base + u1) * Dc + d]) * inv;
    out[(((size_t)b * Sc + s) * Hc + h) * Dc + d] = o;   // [B,S,H*D]
}

// ---------------- launch ----------------
extern "C" void kernel_launch(void* const* d_in, const int* in_sizes, int n_in,
                              void* d_out, int out_size, void* d_ws, size_t ws_size,
                              hipStream_t stream) {
    (void)in_sizes; (void)n_in; (void)out_size; (void)ws_size;
    const float* dec = (const float*)d_in[0];
    const float* hid = (const float*)d_in[1];
    const float* wqk = (const float*)d_in[2];
    const float* wv  = (const float*)d_in[3];
    float* out = (float*)d_out;

    char* ws = (char*)d_ws;
    size_t off = 0;
    auto take = [&](size_t bytes) -> char* {
        char* p = ws + off;
        off += (bytes + 255) & ~(size_t)255;
        return p;
    };
    const size_t BHSD = (size_t)Bc * Hc * Sc * Dc;     // 8.4M elems
    const size_t BHNS = (size_t)Bc * Hc * NSc;         // 262144 elems

    unsigned short* qb  = (unsigned short*)take(BHSD * 2);
    unsigned short* kb  = (unsigned short*)take(BHSD * 2);
    unsigned short* vb  = (unsigned short*)take(BHSD * 2);
    unsigned short* rotbf = (unsigned short*)take((size_t)Hc * 128 * 64 * 2);
    int*   buckets      = (int*)take(2 * BHNS * 4);    // q then k
    int*   sq           = (int*)take(BHNS * 4);
    int*   sk           = (int*)take(BHNS * 4);
    int*   undo         = (int*)take(BHNS * 4);
    float* outws        = (float*)take(BHNS * Dc * 4);
    float* logws        = (float*)take(BHNS * 4);

    rot_kernel<<<dim3((Hc * 128 * 64) / 256), 256, 0, stream>>>(rotbf);
    proj_kernel<<<dim3((Bc * Sc) / 128, Nc / 16, 3), 256, 0, stream>>>(
        dec, hid, wqk, wv, qb, kb, vb);
    hash_kernel<<<dim3(Bc * Hc, Sc / 64, 2), 256, 0, stream>>>(qb, kb, rotbf, buckets);
    sort_kernel<<<dim3(Bc * Hc, 2), 1024, 0, stream>>>(buckets, sq, sk, undo);
    norm_k_kernel<<<dim3((Bc * Hc * Sc) / 8), 256, 0, stream>>>(kb);
    attn_kernel<<<dim3(NCc, Bc * Hc), 256, 0, stream>>>(qb, kb, vb, sq, sk, outws, logws);
    merge_kernel<<<dim3(Sc, Bc * Hc), 64, 0, stream>>>(outws, logws, undo, out);
}
